// Network_1760936591970
// MI455X (gfx1250) — matrix operand order repaired and checked by reading, not compiled, not run on hardware
//
#include <hip/hip_runtime.h>
#include <hip/hip_bf16.h>
#include <math.h>
#include <stdint.h>

// ---------------------------------------------------------------------------
// Fused 3-layer MLP (256 -> 512 -> 512 -> 256) + per-node activation + softmax
// MI455X / gfx1250: bf16 WMMA (v_wmma_f32_16x16x32_bf16), fp32 epilogue with
// hardware v_tanh_f32/v_sin_f32, TDM (tensor_load_to_lds) input staging with
// hardware LDS padding.
// ---------------------------------------------------------------------------

typedef __attribute__((ext_vector_type(16))) __bf16 bf16x16;
typedef __attribute__((ext_vector_type(8)))  __bf16 bf16x8;
typedef __attribute__((ext_vector_type(8)))  float  f32x8;
typedef __attribute__((ext_vector_type(4)))  unsigned int uint32x4;
typedef __attribute__((ext_vector_type(8)))  int int32x8;
typedef __attribute__((ext_vector_type(4)))  int int32x4;

#define BATCH_TILE 64    // batch rows per workgroup
#define TPB        256   // 8 waves of 32
#define LDA_IN     264   // bf16 row stride for 256-wide activations (16B pad)
#define LDA_HID    520   // bf16 row stride for 512-wide activations (16B pad)
#define LDH        260   // fp32 row stride for 256-wide staging / layer-3 out

// ---------------------------------------------------------------------------
// Prep: W[k][n] (fp32, row-major K x N)  ->  WT[n][k] (bf16, row-major N x K)
// ---------------------------------------------------------------------------
__global__ void wt_transpose_bf16(const float* __restrict__ W,
                                  __bf16* __restrict__ WT, int K, int N) {
  int idx = blockIdx.x * blockDim.x + threadIdx.x;  // index into WT (N*K)
  int total = K * N;
  if (idx >= total) return;
  int n = idx / K;
  int k = idx - n * K;
  WT[idx] = (__bf16)W[(size_t)k * N + n];
}

// ---------------------------------------------------------------------------
// hardware tanh (V_TANH_F32 transcendental on gfx1250)
// ---------------------------------------------------------------------------
__device__ inline float fast_tanh(float x) {
#if __has_builtin(__builtin_amdgcn_tanhf)
  return __builtin_amdgcn_tanhf(x);
#else
  return tanhf(x);
#endif
}

// ---------------------------------------------------------------------------
// TDM: 2D tile load, global -> LDS, with per-row LDS padding.
//   elemCode: 0=1B,1=2B,2=4B ; padInterval: pad every 2^padInterval 8B units
//   padAmount: (padAmount+1) DWORDs inserted
// amdgpu-toolchain (clang-23) 6-arg builtin:
//   (uint32x4 g0, int32x8 g1, int32x4 g2, int32x4 g3, int32x8 gx, i32 cpol)
// ---------------------------------------------------------------------------
#if __has_builtin(__builtin_amdgcn_tensor_load_to_lds)
#define USE_TDM 1
__device__ inline void tdm_load_2d_padded(const void* gptr, uint32_t ldsAddr,
                                          uint32_t elemCode, uint32_t tdim0,
                                          uint32_t tdim1, uint32_t tile0,
                                          uint32_t tile1, uint64_t stride0,
                                          uint32_t padInterval,
                                          uint32_t padAmount) {
  uint64_t ga = (uint64_t)(uintptr_t)gptr & ((1ull << 57) - 1ull);
  // D# group 0: count=1 | lds_addr | global_addr[56:0] | type=2
  uint32x4 g0;
  g0.x = 1u;
  g0.y = ldsAddr;
  g0.z = (uint32_t)ga;
  g0.w = (uint32_t)(ga >> 32) | (2u << 30);
  // D# group 1: data_size, pad ctl, tensor/tile dims, dim0 stride
  int32x8 g1;
  g1[0] = (int)((elemCode << 16) | (1u << 20) | (padInterval << 22) |
                (padAmount << 25));
  g1[1] = (int)((tdim0 & 0xFFFFu) << 16);                       // abar=0
  g1[2] = (int)(((tdim0 >> 16) & 0xFFFFu) | ((tdim1 & 0xFFFFu) << 16));
  g1[3] = (int)(((tdim1 >> 16) & 0xFFFFu) | (tile0 << 16));
  g1[4] = (int)tile1;                                           // tile_dim2=0
  g1[5] = (int)(uint32_t)stride0;
  g1[6] = (int)(uint32_t)((stride0 >> 32) & 0xFFFFu);           // stride1=0
  g1[7] = 0;
  int32x4 z4 = {};
  int32x8 z8 = {};
  __builtin_amdgcn_tensor_load_to_lds(g0, g1, z4, z4, z8, 0);
}
#endif

// ---------------------------------------------------------------------------
// WMMA fragment helpers (ISA 7.12.2 layouts)
// ---------------------------------------------------------------------------
__device__ inline bf16x16 join16(bf16x8 lo, bf16x8 hi) {
  return __builtin_shufflevector(lo, hi, 0, 1, 2, 3, 4, 5, 6, 7,
                                         8, 9, 10, 11, 12, 13, 14, 15);
}

// A (16x32, MxK): two contiguous 8-elem chunks at [kb+8*half], [kb+16+8*half]
__device__ inline bf16x16 load_a_frag(const __bf16* rowPtr, int kb) {
  bf16x8 lo = *(const bf16x8*)(rowPtr + kb);
  bf16x8 hi = *(const bf16x8*)(rowPtr + kb + 16);
  return join16(lo, hi);
}

// B (32x16, KxN): lane holds col n = lane%16, two 8-elem chunks at [kb+8*half], [kb+16+8*half]
__device__ inline bf16x16 load_b_frag(const __bf16* colPtr, int kb) {
  bf16x8 lo = *(const bf16x8*)(colPtr + kb);
  bf16x8 hi = *(const bf16x8*)(colPtr + kb + 16);
  return join16(lo, hi);
}

// epilogue of one 16x16 f32 tile: VGPR r <-> row (rowT + r + 8*half), col n
template <bool LAST>
__device__ inline void epilogue_tile(f32x8 acc, int n,
                                     const float* __restrict__ ap,
                                     const float* __restrict__ bias,
                                     __bf16* __restrict__ bufOut, int ldo,
                                     float* __restrict__ fOut,
                                     int rowT, int half) {
  const float* a = ap + (size_t)n * 5;
  float a0 = a[0], a1 = a[1], a2 = a[2], a3 = a[3], a4 = a[4];
  float bn = bias[n];
#pragma unroll
  for (int r = 0; r < 8; ++r) {
    float v = acc[r] + bn;
    float act = a0 * fast_tanh(v) * __sinf(fmaf(a1, v, a2)) + fmaf(a3, v, a4);
    int row = rowT + r + 8 * half;
    if (LAST)
      fOut[row * LDH + n] = act;
    else
      bufOut[row * ldo + n] = (__bf16)act;
  }
}

// ---------------------------------------------------------------------------
// One layer, 2x2 register blocking per wave:
//  wave = (wr, wc); wr in {0,1} -> rows [32*wr, 32*wr+32) as two 16-row tiles,
//  wc in {0..3} -> 32 cols per 128-col chunk as two 16-col tiles.
//  B fragments reused across the two row tiles; A across the two col tiles.
// ---------------------------------------------------------------------------
template <int K, int NOUT, bool LAST>
__device__ inline void layer_gemm(const __bf16* __restrict__ wt,
                                  const float* __restrict__ ap,
                                  const float* __restrict__ bias,
                                  const __bf16* __restrict__ bufIn, int ldi,
                                  __bf16* __restrict__ bufOut, int ldo,
                                  float* __restrict__ fOut,
                                  int wr, int wc, int m16, int half) {
  const int rowT0 = wr * 32;
  const int rowT1 = rowT0 + 16;
  const __bf16* aRow0 = bufIn + (rowT0 + m16) * ldi + 8 * half;
  const __bf16* aRow1 = bufIn + (rowT1 + m16) * ldi + 8 * half;

  for (int c0 = 0; c0 < NOUT; c0 += 128) {
    const int nBase = c0 + wc * 32;
    const int n0 = nBase + m16;
    const int n1 = nBase + 16 + m16;
    const __bf16* bCol0 = wt + (size_t)n0 * K + 8 * half;
    const __bf16* bCol1 = wt + (size_t)n1 * K + 8 * half;

    if (c0 + 128 < NOUT)  // warm next chunk's weight columns
      __builtin_prefetch(wt + (size_t)(n0 + 128) * K, 0, 1);

    f32x8 acc00 = {}, acc01 = {}, acc10 = {}, acc11 = {};
#pragma unroll
    for (int kb = 0; kb < K; kb += 32) {
      bf16x16 a0 = load_a_frag(aRow0, kb);
      bf16x16 a1 = load_a_frag(aRow1, kb);
      bf16x16 b0 = load_b_frag(bCol0, kb);
      bf16x16 b1 = load_b_frag(bCol1, kb);
      acc00 = __builtin_amdgcn_wmma_f32_16x16x32_bf16(false, a0, false, b0,
                                                      (short)0, acc00, false,
                                                      false);
      acc01 = __builtin_amdgcn_wmma_f32_16x16x32_bf16(false, a0, false, b1,
                                                      (short)0, acc01, false,
                                                      false);
      acc10 = __builtin_amdgcn_wmma_f32_16x16x32_bf16(false, a1, false, b0,
                                                      (short)0, acc10, false,
                                                      false);
      acc11 = __builtin_amdgcn_wmma_f32_16x16x32_bf16(false, a1, false, b1,
                                                      (short)0, acc11, false,
                                                      false);
    }
    epilogue_tile<LAST>(acc00, n0, ap, bias, bufOut, ldo, fOut, rowT0, half);
    epilogue_tile<LAST>(acc01, n1, ap, bias, bufOut, ldo, fOut, rowT0, half);
    epilogue_tile<LAST>(acc10, n0, ap, bias, bufOut, ldo, fOut, rowT1, half);
    epilogue_tile<LAST>(acc11, n1, ap, bias, bufOut, ldo, fOut, rowT1, half);
  }
}

// ---------------------------------------------------------------------------
// Fused kernel: one workgroup = 64 batch rows through all 3 layers + softmax
// ---------------------------------------------------------------------------
__global__ __launch_bounds__(TPB) void mlp_fused(
    const float* __restrict__ data,
    const __bf16* __restrict__ wt1, const __bf16* __restrict__ wt2,
    const __bf16* __restrict__ wt3,
    const float* __restrict__ b1, const float* __restrict__ b2,
    const float* __restrict__ b3,
    const float* __restrict__ a1, const float* __restrict__ a2,
    const float* __restrict__ a3,
    float* __restrict__ out) {
  // bufA / bufB: 64 x 520 bf16 each (66560 B).  The fp32 input staging area
  // and the fp32 layer-3 output (64 x 260) both alias bufB.
  __shared__ __align__(16) unsigned char smem[2 * BATCH_TILE * LDA_HID * 2];
  __bf16* bufA = (__bf16*)smem;
  __bf16* bufB = (__bf16*)(smem + BATCH_TILE * LDA_HID * 2);
  float* h3 = (float*)(smem + BATCH_TILE * LDA_HID * 2);

  const int tid = threadIdx.x;
  const int wave = tid >> 5;
  const int lane = tid & 31;
  const int wr = wave & 1;   // row half  (2 x 32 rows)
  const int wc = wave >> 1;  // col group (4 x 32 cols)
  const int m16 = lane & 15;
  const int half = lane >> 4;
  const int blockRow = blockIdx.x * BATCH_TILE;

  // ---- stage input tile (fp32) into LDS -----------------------------------
  const float* src = data + (size_t)blockRow * 256;
#if defined(USE_TDM)
  if (wave == 0) {
    // 64 x 256 fp32 tile; row = 1024 B = 128 x 8B units -> pad_interval 7,
    // pad 4 DWORDs (16 B) per row -> stride-260 fp32 layout in LDS.
    tdm_load_2d_padded(src, (uint32_t)(uintptr_t)h3, /*elem=4B*/ 2,
                       /*tdim0=*/256, /*tdim1=*/BATCH_TILE,
                       /*tile0=*/256, /*tile1=*/BATCH_TILE,
                       /*stride0=*/256, /*padInterval=*/7, /*padAmount=*/3);
    __builtin_amdgcn_s_wait_tensorcnt(0);
  }
  __syncthreads();
  // convert fp32 staging (stride 260) -> bf16 bufA (stride 264)
#pragma unroll
  for (int i = 0; i < (BATCH_TILE * 256) / TPB; ++i) {
    int idx = tid + i * TPB;
    int r = idx >> 8;
    int c = idx & 255;
    bufA[r * LDA_IN + c] = (__bf16)h3[r * LDH + c];
  }
#else
  // fallback: direct global -> bf16 LDS staging
#pragma unroll
  for (int i = 0; i < (BATCH_TILE * 256) / TPB; ++i) {
    int idx = tid + i * TPB;
    int r = idx >> 8;
    int c = idx & 255;
    bufA[r * LDA_IN + c] = (__bf16)src[idx];
  }
#endif
  __syncthreads();

  // ---- layer 1: 256 -> 512 (bufA -> bufB) ---------------------------------
  layer_gemm<256, 512, false>(wt1, a1, b1, bufA, LDA_IN, bufB, LDA_HID,
                              nullptr, wr, wc, m16, half);
  __syncthreads();
  // ---- layer 2: 512 -> 512 (bufB -> bufA) ---------------------------------
  layer_gemm<512, 512, false>(wt2, a2, b2, bufB, LDA_HID, bufA, LDA_HID,
                              nullptr, wr, wc, m16, half);
  __syncthreads();
  // ---- layer 3: 512 -> 256 (bufA -> h3 fp32) ------------------------------
  layer_gemm<512, 256, true>(wt3, a3, b3, bufA, LDA_HID, nullptr, 0, h3, wr,
                             wc, m16, half);
  __syncthreads();

  // ---- softmax over 256 cols per row: 4 lanes cooperate per row -----------
  {
    const int row = tid >> 2;  // 0..63
    const int sub = tid & 3;   // 0..3 within one wave
    float v[64];
#pragma unroll
    for (int j = 0; j < 64; ++j) v[j] = h3[row * LDH + sub + 4 * j];

    float m = -INFINITY;
#pragma unroll
    for (int j = 0; j < 64; ++j) m = fmaxf(m, v[j]);
#pragma unroll
    for (int off = 2; off > 0; off >>= 1) m = fmaxf(m, __shfl_xor(m, off, 4));

    float s = 0.0f;
#pragma unroll
    for (int j = 0; j < 64; ++j) {
      v[j] = __expf(v[j] - m);
      s += v[j];
    }
#pragma unroll
    for (int off = 2; off > 0; off >>= 1) s += __shfl_xor(s, off, 4);

    float inv = 1.0f / s;
    float* dst = out + ((size_t)blockRow + row) * 256;
#pragma unroll
    for (int j = 0; j < 64; ++j) dst[sub + 4 * j] = v[j] * inv;
  }
}

// ---------------------------------------------------------------------------
// launch
// ---------------------------------------------------------------------------
extern "C" void kernel_launch(void* const* d_in, const int* in_sizes, int n_in,
                              void* d_out, int out_size, void* d_ws,
                              size_t ws_size, hipStream_t stream) {
  (void)in_sizes; (void)n_in; (void)out_size; (void)ws_size;

  const float* data = (const float*)d_in[0];
  const float* W1 = (const float*)d_in[1];
  const float* b1 = (const float*)d_in[2];
  const float* a1 = (const float*)d_in[3];
  const float* W2 = (const float*)d_in[4];
  const float* b2 = (const float*)d_in[5];
  const float* a2 = (const float*)d_in[6];
  const float* W3 = (const float*)d_in[7];
  const float* b3 = (const float*)d_in[8];
  const float* a3 = (const float*)d_in[9];
  float* out = (float*)d_out;

  // workspace: WT1 [512x256] | WT2 [512x512] | WT3 [256x512] (bf16)
  unsigned char* ws = (unsigned char*)d_ws;
  __bf16* wt1 = (__bf16*)(ws);                  // 262144 B
  __bf16* wt2 = (__bf16*)(ws + 262144);         // 524288 B
  __bf16* wt3 = (__bf16*)(ws + 262144 + 524288);

  wt_transpose_bf16<<<(256 * 512 + 255) / 256, 256, 0, stream>>>(W1, wt1, 256, 512);
  wt_transpose_bf16<<<(512 * 512 + 255) / 256, 256, 0, stream>>>(W2, wt2, 512, 512);
  wt_transpose_bf16<<<(512 * 256 + 255) / 256, 256, 0, stream>>>(W3, wt3, 512, 256);

  const int B = 65536;
  mlp_fused<<<B / BATCH_TILE, TPB, 0, stream>>>(data, wt1, wt2, wt3, b1, b2,
                                                b3, a1, a2, a3, out);
}
